// Spiking_dend_Self_Attention_12953621364902
// MI455X (gfx1250) — matrix operand order repaired and checked by reading, not compiled, not run on hardware
//
#include <hip/hip_runtime.h>

typedef __bf16 bf16_t;
typedef __attribute__((ext_vector_type(16))) __bf16 v16bf;
typedef __attribute__((ext_vector_type(8)))  __bf16 v8bf;
typedef __attribute__((ext_vector_type(8)))  float  v8f;

#define TT 4
#define BB 32
#define CC 512
#define NN 256      // H*W
#define HEADS 8
#define DDIM 64

__device__ __forceinline__ v8f wmma_bf16(v16bf a, v16bf b, v8f c) {
  return __builtin_amdgcn_wmma_f32_16x16x32_bf16(false, a, false, b, (short)0, c, false, false);
}

// ---------------- f32 -> bf16 weight convert ----------------
__global__ void cvt_bf16_kernel(const float* __restrict__ in, bf16_t* __restrict__ out, int n) {
  int i = blockIdx.x * blockDim.x + threadIdx.x;
  if (i < n) out[i] = (bf16_t)in[i];
}

// ------------- x [tb][c][n] f32 -> xT [tb][n][c] bf16 -------------
__global__ void transpose_cn_kernel(const float* __restrict__ in, bf16_t* __restrict__ out) {
  int gid = blockIdx.x * 256 + threadIdx.x;          // < T*B*C*N
  int n  = gid & (NN - 1);
  int c  = (gid >> 8) & (CC - 1);
  int tb = gid >> 17;                                // / (C*N)
  out[(size_t)tb * (NN * CC) + (size_t)n * CC + c] = (bf16_t)in[gid];
}

// ---------------- 1x1 conv as GEMM: out[tb][o][n] = sum_c W[o][c] * X[tb][n][c] ----------------
// block tile M=64 N=128, 8 waves, each wave 32x32 (4 WMMA tiles), K=512
__global__ __launch_bounds__(256) void conv_gemm_kernel(
    const bf16_t* __restrict__ W,      // [512][512], c contiguous
    const bf16_t* __restrict__ X,      // [TB][256][512], c contiguous
    const float* __restrict__ scale, const float* __restrict__ bias,
    const float* __restrict__ pb, int has_pb,
    float* __restrict__ out)           // [TB][512][256]
{
  const int tb    = blockIdx.x;
  const int m_blk = blockIdx.y * 64;
  const int n_blk = blockIdx.z * 128;
  const int wid   = threadIdx.x >> 5;
  const int lane  = threadIdx.x & 31;
  const int lrow  = lane & 15, lhalf = lane >> 4;
  const int m0 = m_blk + (wid & 1) * 32;
  const int n0 = n_blk + (wid >> 1) * 32;

  const bf16_t* Xb = X + (size_t)tb * NN * CC;
  v8f acc00 = {}, acc01 = {}, acc10 = {}, acc11 = {};

  for (int k0 = 0; k0 < CC; k0 += 32) {
    v16bf a0, a1, b0, b1;
    const bf16_t* pa0 = W + (size_t)(m0 + lrow) * CC + k0;
    ((v8bf*)&a0)[0] = *(const v8bf*)(pa0 + lhalf * 8);
    ((v8bf*)&a0)[1] = *(const v8bf*)(pa0 + 16 + lhalf * 8);
    const bf16_t* pa1 = W + (size_t)(m0 + 16 + lrow) * CC + k0;
    ((v8bf*)&a1)[0] = *(const v8bf*)(pa1 + lhalf * 8);
    ((v8bf*)&a1)[1] = *(const v8bf*)(pa1 + 16 + lhalf * 8);

    const bf16_t* pb0 = Xb + (size_t)(n0 + lrow) * CC + k0;
    ((v8bf*)&b0)[0] = *(const v8bf*)(pb0 + lhalf * 8);
    ((v8bf*)&b0)[1] = *(const v8bf*)(pb0 + 16 + lhalf * 8);
    const bf16_t* pb1 = Xb + (size_t)(n0 + 16 + lrow) * CC + k0;
    ((v8bf*)&b1)[0] = *(const v8bf*)(pb1 + lhalf * 8);
    ((v8bf*)&b1)[1] = *(const v8bf*)(pb1 + 16 + lhalf * 8);

    acc00 = wmma_bf16(a0, b0, acc00);
    acc01 = wmma_bf16(a0, b1, acc01);
    acc10 = wmma_bf16(a1, b0, acc10);
    acc11 = wmma_bf16(a1, b1, acc11);
  }

  float* ob = out + (size_t)tb * CC * NN;
  auto store_tile = [&](const v8f& acc, int mt, int nt) {
    #pragma unroll
    for (int r = 0; r < 8; ++r) {
      int m = mt + r + 8 * lhalf;
      int n = nt + lrow;
      float v = acc[r];
      if (has_pb) v += pb[m];
      v = v * scale[m] + bias[m];
      ob[(size_t)m * NN + n] = v;
    }
  };
  store_tile(acc00, m0,      n0);
  store_tile(acc01, m0,      n0 + 16);
  store_tile(acc10, m0 + 16, n0);
  store_tile(acc11, m0 + 16, n0 + 16);
}

// ---------------- multi-step LIF over T; MODE 0: bf16 same layout, 1: bf16 [n][c]-transposed, 2: f32 same layout ----------------
template<int MODE>
__global__ void lif_kernel(const float* __restrict__ u, void* __restrict__ out, float vth) {
  const int S = BB * CC * NN;
  int gid = blockIdx.x * 256 + threadIdx.x;
  int n  = gid & (NN - 1);
  int c  = (gid >> 8) & (CC - 1);
  int tb = gid >> 17;
  size_t tr = (size_t)tb * (NN * CC) + (size_t)n * CC + c;
  float v = 0.0f;
  #pragma unroll
  for (int t = 0; t < TT; ++t) {
    float x = u[(size_t)t * S + gid];
    v = 0.5f * (v + x);                 // v += (x - v)/tau, tau = 2
    float s = (v >= vth) ? 1.0f : 0.0f;
    v = (s != 0.0f) ? 0.0f : v;         // hard reset
    if (MODE == 0)      ((bf16_t*)out)[(size_t)t * S + gid] = (bf16_t)s;
    else if (MODE == 1) ((bf16_t*)out)[(size_t)t * S + tr]  = (bf16_t)s;
    else                ((float*)out)[(size_t)t * S + gid]  = s;
  }
}

// ---------------- attention: per (t,b,h) block: kv = K^T V (LDS), a = Q kv * 0.125 ----------------
__global__ __launch_bounds__(256) void attn_kernel(
    const bf16_t* __restrict__ Qt,   // [TB][256][512]  (n-major, c contiguous)
    const bf16_t* __restrict__ Ks,   // [TB][512][256]  (c-major, n contiguous) == K^T per head
    const bf16_t* __restrict__ Vs,   // [TB][512][256]
    float* __restrict__ out)         // [TB][512][256] : out[c=h*64+e][n]
{
  __shared__ __align__(16) bf16_t kvT[64][72];   // kvT[e][dd] = kv[dd][e] * 0.125
  const int h  = blockIdx.x & 7;
  const int tb = blockIdx.x >> 3;
  const int wid  = threadIdx.x >> 5;
  const int lane = threadIdx.x & 31;
  const int lrow = lane & 15, lhalf = lane >> 4;

  const bf16_t* Kb = Ks + (size_t)tb * CC * NN + (size_t)(h * DDIM) * NN;
  const bf16_t* Vb = Vs + (size_t)tb * CC * NN + (size_t)(h * DDIM) * NN;

  // Stage 1: kv (64x64), K-dim = n = 256. 16 tiles over 8 waves (2 each).
  {
    const int mi  = wid >> 1;            // dd tile 0..3
    const int ni0 = (wid & 1) * 2;       // e tiles ni0, ni0+1
    v8f acc0 = {}, acc1 = {};
    for (int k0 = 0; k0 < NN; k0 += 32) {
      v16bf a, b0, b1;
      const bf16_t* pa = Kb + (size_t)(mi * 16 + lrow) * NN + k0;
      ((v8bf*)&a)[0] = *(const v8bf*)(pa + lhalf * 8);
      ((v8bf*)&a)[1] = *(const v8bf*)(pa + 16 + lhalf * 8);
      const bf16_t* p0 = Vb + (size_t)(ni0 * 16 + lrow) * NN + k0;
      ((v8bf*)&b0)[0] = *(const v8bf*)(p0 + lhalf * 8);
      ((v8bf*)&b0)[1] = *(const v8bf*)(p0 + 16 + lhalf * 8);
      const bf16_t* p1 = Vb + (size_t)((ni0 + 1) * 16 + lrow) * NN + k0;
      ((v8bf*)&b1)[0] = *(const v8bf*)(p1 + lhalf * 8);
      ((v8bf*)&b1)[1] = *(const v8bf*)(p1 + 16 + lhalf * 8);
      acc0 = wmma_bf16(a, b0, acc0);
      acc1 = wmma_bf16(a, b1, acc1);
    }
    #pragma unroll
    for (int r = 0; r < 8; ++r) {
      int dd = mi * 16 + r + 8 * lhalf;
      kvT[ni0 * 16 + lrow][dd]       = (bf16_t)(acc0[r] * 0.125f);
      kvT[(ni0 + 1) * 16 + lrow][dd] = (bf16_t)(acc1[r] * 0.125f);
    }
  }
  __syncthreads();

  // Stage 2: a = Q @ kv : M=256 (n), K=64 (dd), N=64 (e). Wave w owns n rows [32w, 32w+32).
  {
    const int nw = wid * 32;
    const bf16_t* Qb = Qt + (size_t)tb * NN * CC + h * DDIM;
    v8f acc[2][4] = {};
    #pragma unroll
    for (int k0 = 0; k0 < DDIM; k0 += 32) {
      v16bf a[2], b[4];
      #pragma unroll
      for (int mi = 0; mi < 2; ++mi) {
        const bf16_t* p = Qb + (size_t)(nw + mi * 16 + lrow) * CC + k0;
        ((v8bf*)&a[mi])[0] = *(const v8bf*)(p + lhalf * 8);
        ((v8bf*)&a[mi])[1] = *(const v8bf*)(p + 16 + lhalf * 8);
      }
      #pragma unroll
      for (int ni = 0; ni < 4; ++ni) {
        const bf16_t* p = &kvT[ni * 16 + lrow][k0];
        ((v8bf*)&b[ni])[0] = *(const v8bf*)(p + lhalf * 8);
        ((v8bf*)&b[ni])[1] = *(const v8bf*)(p + 16 + lhalf * 8);
      }
      #pragma unroll
      for (int mi = 0; mi < 2; ++mi)
        #pragma unroll
        for (int ni = 0; ni < 4; ++ni)
          acc[mi][ni] = wmma_bf16(a[mi], b[ni], acc[mi][ni]);
    }
    float* ob = out + (size_t)tb * CC * NN + (size_t)(h * DDIM) * NN;
    #pragma unroll
    for (int mi = 0; mi < 2; ++mi)
      #pragma unroll
      for (int ni = 0; ni < 4; ++ni)
        #pragma unroll
        for (int r = 0; r < 8; ++r) {
          int n = nw + mi * 16 + r + 8 * lhalf;
          int e = ni * 16 + lrow;
          ob[(size_t)e * NN + n] = acc[mi][ni][r];
        }
  }
}

extern "C" void kernel_launch(void* const* d_in, const int* in_sizes, int n_in,
                              void* d_out, int out_size, void* d_ws, size_t ws_size,
                              hipStream_t stream) {
  const float* x   = (const float*)d_in[0];
  const float* qw  = (const float*)d_in[1];
  const float* qsc = (const float*)d_in[2];
  const float* qbi = (const float*)d_in[3];
  const float* kw  = (const float*)d_in[4];
  const float* ksc = (const float*)d_in[5];
  const float* kbi = (const float*)d_in[6];
  const float* vw  = (const float*)d_in[7];
  const float* vsc = (const float*)d_in[8];
  const float* vbi = (const float*)d_in[9];
  const float* pw  = (const float*)d_in[10];
  const float* pbv = (const float*)d_in[11];
  const float* psc = (const float*)d_in[12];
  const float* pbi = (const float*)d_in[13];

  char* ws = (char*)d_ws;
  // workspace layout (bytes):
  //   [0,   32MB)   xT bf16 [tb][n][c]   (reused later for attn-spike transposed s2T)
  //   [32MB,34MB)   4 bf16 weights
  //   [34MB, 98MB) u f32 [tb][c][n] (conv / attention outputs, pre-LIF)
  //   then qT / k / v spike tensors, 32MB each
  bf16_t* xT  = (bf16_t*)(ws);
  bf16_t* wqB = (bf16_t*)(ws + 33554432);
  bf16_t* wkB = wqB + 262144;
  bf16_t* wvB = wkB + 262144;
  bf16_t* wpB = wvB + 262144;
  float*  u   = (float*)(ws + 35651584);
  bf16_t* qT  = (bf16_t*)(ws + 102760448);
  bf16_t* kS  = (bf16_t*)(ws + 136314880);
  bf16_t* vS  = (bf16_t*)(ws + 169869312);

  const int NTOT = TT * BB * CC * NN;       // 16,777,216
  const int SLIF = (BB * CC * NN) / 256;    // 16384 blocks

  cvt_bf16_kernel<<<1024, 256, 0, stream>>>(qw, wqB, 262144);
  cvt_bf16_kernel<<<1024, 256, 0, stream>>>(kw, wkB, 262144);
  cvt_bf16_kernel<<<1024, 256, 0, stream>>>(vw, wvB, 262144);
  cvt_bf16_kernel<<<1024, 256, 0, stream>>>(pw, wpB, 262144);
  transpose_cn_kernel<<<NTOT / 256, 256, 0, stream>>>(x, xT);

  dim3 cg(TT * BB, CC / 64, NN / 128);
  // Q path (spikes stored [n][c]-transposed for stage-2 A fragments)
  conv_gemm_kernel<<<cg, 256, 0, stream>>>(wqB, xT, qsc, qbi, nullptr, 0, u);
  lif_kernel<1><<<SLIF, 256, 0, stream>>>(u, qT, 1.0f);
  // K path (kept [c][n] == K^T layout for stage-1 A fragments)
  conv_gemm_kernel<<<cg, 256, 0, stream>>>(wkB, xT, ksc, kbi, nullptr, 0, u);
  lif_kernel<0><<<SLIF, 256, 0, stream>>>(u, kS, 1.0f);
  // V path
  conv_gemm_kernel<<<cg, 256, 0, stream>>>(wvB, xT, vsc, vbi, nullptr, 0, u);
  lif_kernel<0><<<SLIF, 256, 0, stream>>>(u, vS, 1.0f);
  // attention (writes a*0.125 into u in [c][n] layout)
  attn_kernel<<<TT * BB * HEADS, 256, 0, stream>>>(qT, kS, vS, u);
  // attn LIF v_th=0.5, spikes transposed into xT region (s2T) for the projection GEMM
  lif_kernel<1><<<SLIF, 256, 0, stream>>>(u, xT, 0.5f);
  // projection conv: (conv + pb) * scale + bias
  conv_gemm_kernel<<<cg, 256, 0, stream>>>(wpB, xT, psc, pbi, pbv, 1, u);
  // final LIF v_th=1.0 -> f32 output [T,B,C,H,W]
  lif_kernel<2><<<SLIF, 256, 0, stream>>>(u, d_out, 1.0f);
}
